// DeformableConv2d_55697135895128
// MI455X (gfx1250) — matrix operand order repaired and checked by reading, not compiled, not run on hardware
//
#include <hip/hip_runtime.h>
#include <hip/hip_bf16.h>

typedef __attribute__((ext_vector_type(16))) __bf16        v16bf;
typedef __attribute__((ext_vector_type(8)))  float         v8f;
typedef __attribute__((ext_vector_type(4)))  unsigned int  v4u;

#define B_    4
#define C_    64
#define H_    160
#define W_    160
#define OC_   64
#define NN_   9          // KS*KS
#define HO_   158
#define WO_   158
#define HW_   (H_*W_)    // 25600
#define HWO_  (HO_*WO_)  // 24964
#define HP_   162        // H + 2*pad
#define TILE_J 32

__device__ __forceinline__ unsigned short f2bf(float f) {
  unsigned u = __float_as_uint(f);
  u += 0x7FFFu + ((u >> 16) & 1u);        // round-to-nearest-even
  return (unsigned short)(u >> 16);
}

// ---------------------------------------------------------------------------
// Stage 1: offset field conv. One thread per (b,i,j), 18 output channels kept
// in registers; w_off (41.5 KB) cached in LDS -> 18-way reuse of each x load.
// ---------------------------------------------------------------------------
__global__ void __launch_bounds__(256)
offset_conv_kernel(const float* __restrict__ x,
                   const float* __restrict__ w_off,
                   const float* __restrict__ b_off,
                   float* __restrict__ off_out) {
  __shared__ float s_w[18 * 576];
  for (int t = threadIdx.x; t < 18 * 576; t += blockDim.x) s_w[t] = w_off[t];
  __syncthreads();

  int p = blockIdx.x * blockDim.x + threadIdx.x;
  if (p >= B_ * HWO_) return;
  int b = p / HWO_;
  int r = p - b * HWO_;
  int i = r / WO_;
  int j = r - i * WO_;

  float acc[18];
#pragma unroll
  for (int ch = 0; ch < 18; ++ch) acc[ch] = b_off[ch];

  const float* xb = x + (size_t)b * C_ * HW_;
  for (int c = 0; c < C_; ++c) {
    const float* xc = xb + c * HW_;
#pragma unroll
    for (int kh = 0; kh < 3; ++kh) {
#pragma unroll
      for (int kw = 0; kw < 3; ++kw) {
        float xv = xc[(i + kh) * W_ + (j + kw)];
        int kidx = kh * 3 + kw;
#pragma unroll
        for (int ch = 0; ch < 18; ++ch)
          acc[ch] = fmaf(xv, s_w[ch * 576 + c * 9 + kidx], acc[ch]);
      }
    }
  }
#pragma unroll
  for (int ch = 0; ch < 18; ++ch)
    off_out[((size_t)b * 18 + ch) * HWO_ + r] = acc[ch];
}

// ---------------------------------------------------------------------------
// Stage 1b: repack w_d (OIHW fp32) -> bf16 table w2[n][oc][c] so A-fragments
// are contiguous 16B loads in the 16-bit 16x32 WMMA A layout.
// ---------------------------------------------------------------------------
__global__ void wconv_kernel(const float* __restrict__ w_d,
                             unsigned short* __restrict__ w2) {
  int t = blockIdx.x * blockDim.x + threadIdx.x;
  if (t >= NN_ * OC_ * C_) return;
  int n  = t / (OC_ * C_);
  int r  = t - n * (OC_ * C_);
  int oc = r / C_;
  int c  = r - oc * C_;
  w2[t] = f2bf(w_d[(oc * C_ + c) * NN_ + n]);
}

// ---------------------------------------------------------------------------
// Stage 2: fused bilinear sampler + implicit GEMM via bf16 WMMA.
// Block = 256 thr = 8 waves; output tile 64(oc) x 32(pos in one output row).
// K loop over the 9 sampling points; each point = 64 channels = 2 WMMA steps.
// ---------------------------------------------------------------------------
__global__ void __launch_bounds__(256)
deform_gemm_kernel(const float* __restrict__ x,
                   const float* __restrict__ offs,
                   const unsigned short* __restrict__ w2,
                   const float* __restrict__ b_d,
                   float* __restrict__ out) {
  __shared__ int   s_idx[4][TILE_J];
  __shared__ float s_g[4][TILE_J];
  __shared__ __align__(32) unsigned short s_col[TILE_J][C_];   // [pos][c] bf16

  const int tid  = threadIdx.x;
  const int lane = tid & 31;
  const int wid  = tid >> 5;

  const int tiles = (WO_ + TILE_J - 1) / TILE_J;   // 5
  int blk = blockIdx.x;
  int tj  = blk % tiles;
  int bi  = blk / tiles;
  int i   = bi % HO_;
  int b   = bi / HO_;
  int j0  = tj * TILE_J;

  const float* xb = x + (size_t)b * C_ * HW_;
  const int mt  = wid & 3;          // 4 M tiles of 16 oc
  const int nt  = wid >> 2;         // 2 N tiles of 16 pos
  const int ocb = mt * 16;
  const int hi  = lane >> 4;
  const int lm  = lane & 15;

  v8f acc = {};

  for (int n = 0; n < NN_; ++n) {
    __syncthreads();
    // --- sampling parameters for the 32 positions (exact reference math) ---
    if (tid < TILE_J) {
      int pos = tid;
      int jj = j0 + pos; if (jj > WO_ - 1) jj = WO_ - 1;
      int sp = i * WO_ + jj;
      float ox = offs[((size_t)b * 18 + 2 * n) * HWO_ + sp];
      float oy = offs[((size_t)b * 18 + 2 * n + 1) * HWO_ + sp];
      float px = ox + (float)(n / 3 - 1) + (float)(i + 1);
      float py = oy + (float)(n % 3 - 1) + (float)(jj + 1);
      float fx = floorf(px), fy = floorf(py);
      int qltx = (int)fminf(fmaxf(fx,        0.f), (float)(HP_ - 1));
      int qlty = (int)fminf(fmaxf(fy,        0.f), (float)(HP_ - 1));
      int qrbx = (int)fminf(fmaxf(fx + 1.f,  0.f), (float)(HP_ - 1));
      int qrby = (int)fminf(fmaxf(fy + 1.f,  0.f), (float)(HP_ - 1));
      float mx = (px < 1.f || px > (float)(HP_ - 2)) ? 1.f : 0.f;
      float my = (py < 1.f || py > (float)(HP_ - 2)) ? 1.f : 0.f;
      px = fminf(fmaxf(px * (1.f - mx) + fx * mx, 0.f), (float)(HP_ - 1));
      py = fminf(fmaxf(py * (1.f - my) + fy * my, 0.f), (float)(HP_ - 1));
      float ltx = (float)qltx, lty = (float)qlty;
      float rbx = (float)qrbx, rby = (float)qrby;
      float glt = (1.f + (ltx - px)) * (1.f + (lty - py));
      float grb = (1.f - (rbx - px)) * (1.f - (rby - py));
      float glb = (1.f + (ltx - px)) * (1.f - (rby - py));
      float grt = (1.f - (rbx - px)) * (1.f + (lty - py));
      int   qx[4] = {qltx, qrbx, qltx, qrbx};
      int   qy[4] = {qlty, qrby, qrby, qlty};
      float g4[4] = {glt,  grb,  glb,  grt};
#pragma unroll
      for (int k = 0; k < 4; ++k) {
        bool valid = (qx[k] >= 1) && (qx[k] <= H_) && (qy[k] >= 1) && (qy[k] <= W_);
        s_idx[k][pos] = valid ? ((qx[k] - 1) * W_ + (qy[k] - 1)) : 0;
        s_g[k][pos]   = valid ? g4[k] : 0.f;
      }
    }
    __syncthreads();
    // --- bilinear gather of this K-chunk (64 channels x 32 positions) ------
    {
      int pos = tid & 31;                // lane == position -> coalesced-ish
      int i0 = s_idx[0][pos], i1 = s_idx[1][pos],
          i2 = s_idx[2][pos], i3 = s_idx[3][pos];
      float g0 = s_g[0][pos], g1 = s_g[1][pos],
            g2 = s_g[2][pos], g3 = s_g[3][pos];
      int cb = tid >> 5;
#pragma unroll
      for (int it = 0; it < 8; ++it) {
        int c = cb + it * 8;
        const float* xc = xb + c * HW_;
        float v = g0 * xc[i0] + g1 * xc[i1] + g2 * xc[i2] + g3 * xc[i3];
        s_col[pos][c] = f2bf(v);
      }
    }
    __syncthreads();
    // --- 2 x v_wmma_f32_16x16x32_bf16 per wave for this K-chunk ------------
    {
      const unsigned short* wrow = w2 + ((size_t)n * OC_ + (ocb + lm)) * C_;
#pragma unroll
      for (int ks2 = 0; ks2 < 2; ++ks2) {
        int kb = ks2 * 32;
        union { v4u q[2]; v16bf v; } A;
        // A 16x32 bf16, contiguous k order used here (matches B): elems e -> K = hi*16 + e
        A.q[0] = *(const v4u*)(wrow + kb + hi * 16);
        A.q[1] = *(const v4u*)(wrow + kb + hi * 16 + 8);
        // B 32x16 bf16, same contiguous k order as A: N = lane&15 ; elems e -> K = (lane>>4)*16 + e
        v16bf Bv = *(const v16bf*)&s_col[nt * 16 + lm][kb + hi * 16];
        acc = __builtin_amdgcn_wmma_f32_16x16x32_bf16(
            false, A.v, false, Bv, (short)0, acc, false, false);
      }
    }
  }

  // --- epilogue: bias + store (C/D layout: VGPR v -> M = v + hi*8) ---------
  int pos = nt * 16 + lm;
  int j = j0 + pos;
  if (j < WO_) {
#pragma unroll
    for (int v = 0; v < 8; ++v) {
      int oc = ocb + hi * 8 + v;
      out[(((size_t)b * OC_ + oc) * HO_ + i) * WO_ + j] = acc[v] + b_d[oc];
    }
  }
}

// ---------------------------------------------------------------------------
extern "C" void kernel_launch(void* const* d_in, const int* in_sizes, int n_in,
                              void* d_out, int out_size, void* d_ws, size_t ws_size,
                              hipStream_t stream) {
  const float* x     = (const float*)d_in[0];
  const float* w_off = (const float*)d_in[1];
  const float* b_off = (const float*)d_in[2];
  const float* w_d   = (const float*)d_in[3];
  const float* b_d   = (const float*)d_in[4];
  float* out = (float*)d_out;

  // workspace: fp32 offset field (7.19 MB) then bf16 weight table (72 KB)
  float* offs = (float*)d_ws;
  size_t offBytes = (size_t)B_ * 18 * HWO_ * sizeof(float);
  unsigned short* w2 = (unsigned short*)((char*)d_ws + offBytes);

  int total1 = B_ * HWO_;
  offset_conv_kernel<<<(total1 + 255) / 256, 256, 0, stream>>>(x, w_off, b_off, offs);

  int total2 = NN_ * OC_ * C_;
  wconv_kernel<<<(total2 + 255) / 256, 256, 0, stream>>>(w_d, w2);

  int tiles = (WO_ + TILE_J - 1) / TILE_J;
  deform_gemm_kernel<<<B_ * HO_ * tiles, 256, 0, stream>>>(x, offs, w2, b_d, out);
}
